// CosFormerAttention_62302795596634
// MI455X (gfx1250) — matrix operand order repaired and checked by reading, not compiled, not run on hardware
//
#include <hip/hip_runtime.h>

typedef __attribute__((ext_vector_type(16))) __bf16 v16bf;
typedef __attribute__((ext_vector_type(8)))  __bf16 v8bf;
typedef __attribute__((ext_vector_type(8)))  float  v8f;
typedef __attribute__((ext_vector_type(4)))  unsigned int u32x4;
typedef __attribute__((ext_vector_type(8)))  int i32x8;
typedef __attribute__((ext_vector_type(4)))  int i32x4;

constexpr int kB  = 2;
constexpr int kT  = 1024;
constexpr int kD  = 512;    // model dim == H*DK
constexpr int kH  = 8;
constexpr int kDK = 64;
constexpr int kN  = kB * kT;     // 2048 rows
constexpr int kC  = 64;          // chunk length
constexpr int kNC = kT / kC;     // 16 chunks
constexpr int kBH = kB * kH;     // 16

#if defined(__has_builtin)
#if __has_builtin(__builtin_amdgcn_tensor_load_to_lds)
#define HAS_TDM 1
#endif
#if __has_builtin(__builtin_amdgcn_s_wait_tensorcnt)
#define HAS_TWAIT 1
#endif
#endif

__device__ __forceinline__ __bf16 f2bf(float f) {
  unsigned u = __float_as_uint(f);
  unsigned r = (u + 0x7FFFu + ((u >> 16) & 1u)) >> 16;
  unsigned short s = (unsigned short)r;
  return __builtin_bit_cast(__bf16, s);
}
__device__ __forceinline__ float bf2f(__bf16 b) {
  unsigned short s = __builtin_bit_cast(unsigned short, b);
  return __uint_as_float(((unsigned)s) << 16);
}

__device__ __forceinline__ v8f wmma_bf16(v16bf a, v16bf b, v8f c) {
  return __builtin_amdgcn_wmma_f32_16x16x32_bf16(false, a, false, b, (short)0, c,
                                                 false, false);
}

// A-tile (16x32 bf16) per-lane fragment (contiguous k = 16*half + i).
__device__ __forceinline__ v16bf load_a16(const __bf16* rowk0, int half) {
  v8bf lo = *(const v8bf*)(rowk0 + 16 * half);
  v8bf hi = *(const v8bf*)(rowk0 + 16 * half + 8);
  v16bf a;
#pragma unroll
  for (int i = 0; i < 8; ++i) { a[i] = lo[i]; a[i + 8] = hi[i]; }
  return a;
}

// low 32 bits of a generic pointer to __shared__ = LDS byte offset
__device__ __forceinline__ unsigned lds_off32(const void* p) {
  return (unsigned)(unsigned long long)p;
}

// ---------------- conversion kernels ----------------
__global__ void k_conv_x(const float* __restrict__ x, __bf16* __restrict__ xb, int n) {
  int i = blockIdx.x * blockDim.x + threadIdx.x;
  if (i < n) xb[i] = f2bf(x[i]);
}

__global__ void k_conv_wT(const float* w0, const float* w1, const float* w2,
                          const float* w3, const float* w4,
                          __bf16* d0, __bf16* d1, __bf16* d2, __bf16* d3, __bf16* d4) {
  const float* src; __bf16* dst;
  switch (blockIdx.y) {
    case 0: src = w0; dst = d0; break;
    case 1: src = w1; dst = d1; break;
    case 2: src = w2; dst = d2; break;
    case 3: src = w3; dst = d3; break;
    default: src = w4; dst = d4; break;
  }
  int i = blockIdx.x * blockDim.x + threadIdx.x;   // over 512*512
  int k = i >> 9, n = i & 511;
  dst[n * 512 + k] = f2bf(src[i]);
}

// ---------------- projection GEMMs: 16x64 tile per wave, double-buffered ----------------
__global__ void k_proj(const __bf16* __restrict__ xb,
                       const __bf16* __restrict__ WqT, const __bf16* __restrict__ WkT,
                       const __bf16* __restrict__ WvT, const __bf16* __restrict__ WgT,
                       const float*  __restrict__ bg,
                       __bf16* __restrict__ Qc, __bf16* __restrict__ Qs,
                       __bf16* __restrict__ Kc, __bf16* __restrict__ Ks,
                       __bf16* __restrict__ Vb, float* __restrict__ G) {
  const int wave = threadIdx.x >> 5;
  const int lane = threadIdx.x & 31;
  const int job  = blockIdx.x * 8 + wave;   // 0..4095: 4 weights x 128 x 8
  const int w    = job >> 10;
  const int rem  = job & 1023;
  const int tm   = rem >> 3;                // 0..127 row tile
  const int tg   = rem & 7;                 // 0..7 (64-wide col group)
  const int half = lane >> 4;
  const int l16  = lane & 15;
  const __bf16* WT = (w == 0) ? WqT : (w == 1) ? WkT : (w == 2) ? WvT : WgT;

  const __bf16* aptr = xb + (size_t)(tm * 16 + l16) * kD;
  const __bf16* bptr = WT + (size_t)(tg * 64 + l16) * kD + 16 * half;

  v8f acc[4] = {};
  // software pipeline: prefetch next K-slice fragments before issuing WMMAs
  v16bf a0 = load_a16(aptr, half);
  v16bf b0[4];
#pragma unroll
  for (int t4 = 0; t4 < 4; ++t4)
    b0[t4] = *(const v16bf*)(bptr + (size_t)t4 * 16 * kD);
#pragma unroll
  for (int k0 = 32; k0 < kD; k0 += 32) {
    v16bf a1 = load_a16(aptr + k0, half);
    v16bf b1[4];
#pragma unroll
    for (int t4 = 0; t4 < 4; ++t4)
      b1[t4] = *(const v16bf*)(bptr + (size_t)t4 * 16 * kD + k0);
#pragma unroll
    for (int t4 = 0; t4 < 4; ++t4)
      acc[t4] = wmma_bf16(a0, b0[t4], acc[t4]);
    a0 = a1;
#pragma unroll
    for (int t4 = 0; t4 < 4; ++t4) b0[t4] = b1[t4];
  }
#pragma unroll
  for (int t4 = 0; t4 < 4; ++t4)
    acc[t4] = wmma_bf16(a0, b0[t4], acc[t4]);

  constexpr float kAng = 3.14159265358979323846f / (2.0f * (float)kT);
#pragma unroll
  for (int ii = 0; ii < 8; ++ii) {
    int r = tm * 16 + ii + 8 * half;
    float t  = (float)(r & (kT - 1));
    float cw = __cosf(t * kAng), sw = __sinf(t * kAng);
#pragma unroll
    for (int t4 = 0; t4 < 4; ++t4) {
      int cc = tg * 64 + t4 * 16 + l16;
      size_t idx = (size_t)r * kD + cc;
      float v = acc[t4][ii];
      if (w <= 1) {
        float rl = v > 0.f ? v : 0.f;
        if (w == 0) { Qc[idx] = f2bf(rl * cw); Qs[idx] = f2bf(rl * sw); }
        else        { Kc[idx] = f2bf(rl * cw); Ks[idx] = f2bf(rl * sw); }
      } else if (w == 2) {
        Vb[idx] = f2bf(v);
      } else {
        G[idx] = 1.f / (1.f + __expf(-(v + bg[cc])));
      }
    }
  }
}

// ---------------- phase A: per-chunk KV sums Ac=Kc^T@V, As=Ks^T@V ----------------
__global__ void k_chunk_kv(const __bf16* __restrict__ Kc, const __bf16* __restrict__ Ks,
                           const __bf16* __restrict__ Vb,
                           float* __restrict__ Ac, float* __restrict__ As,
                           float* __restrict__ kcs, float* __restrict__ kss) {
  __shared__ __bf16 sKc[kC * 72];      // 72-elem stride (TDM pad: 4 dw / 32 dw)
  __shared__ __bf16 sKs[kC * 72];
  __shared__ __bf16 sVt[kDK * 80];
  const int blk = blockIdx.x;          // bh*16 + c, 256 blocks
  const int bh = blk >> 4, c = blk & 15;
  const int b = bh >> 3, h = bh & 7;
  const int row0 = b * kT + c * kC;
  const int col0 = h * kDK;
  const int tid = threadIdx.x;

#ifdef HAS_TDM
  if (tid == 0) {
    // D# group1: data_size=2B, pad_enable, pad_interval=32dw, pad_amount=4dw,
    // tensor_dim0=64, tensor_dim1=64, tile 64x64, dim0_stride=512
    unsigned g1w0 = (1u << 16) | (1u << 20) | (4u << 22) | (3u << 25);
    i32x8 g1 = {(int)g1w0, (int)(64u << 16), (int)(64u << 16), (int)(64u << 16),
                64, 512, 0, 0};
    i32x4 z4 = {0, 0, 0, 0};
    i32x8 z8 = {0, 0, 0, 0, 0, 0, 0, 0};
    unsigned long long ga = (unsigned long long)(Kc + (size_t)row0 * kD + col0);
    u32x4 g0 = {1u, lds_off32(&sKc[0]), (unsigned)ga,
                (unsigned)((ga >> 32) & 0x1FFFFFFu) | (2u << 30)};
    __builtin_amdgcn_tensor_load_to_lds(g0, g1, z4, z4, z8, 0);
    ga = (unsigned long long)(Ks + (size_t)row0 * kD + col0);
    u32x4 g0b = {1u, lds_off32(&sKs[0]), (unsigned)ga,
                 (unsigned)((ga >> 32) & 0x1FFFFFFu) | (2u << 30)};
    __builtin_amdgcn_tensor_load_to_lds(g0b, g1, z4, z4, z8, 0);
  }
#else
#pragma unroll
  for (int i = 0; i < 2; ++i) {
    int ci = tid + i * 256;            // 0..511 : 16B chunks of 64x64
    int t = ci >> 3, d8 = (ci & 7) * 8;
    *(v8bf*)&sKc[t * 72 + d8] = *(const v8bf*)(Kc + (size_t)(row0 + t) * kD + col0 + d8);
    *(v8bf*)&sKs[t * 72 + d8] = *(const v8bf*)(Ks + (size_t)(row0 + t) * kD + col0 + d8);
  }
#endif
  // V staged transposed (manual)
#pragma unroll
  for (int i = 0; i < 16; ++i) {
    int li = tid + i * 256;
    int t = li >> 6, d = li & 63;
    sVt[d * 80 + t] = Vb[(size_t)(row0 + t) * kD + col0 + d];
  }
#ifdef HAS_TWAIT
  __builtin_amdgcn_s_wait_tensorcnt(0);
#endif
  __syncthreads();

  const int wave = tid >> 5, lane = tid & 31;
  const int half = lane >> 4, l16 = lane & 15;
  for (int j = wave; j < 32; j += 8) {      // 16 tiles x 2 states
    int state = j >> 4;
    int dt = (j >> 2) & 3, et = j & 3;
    const __bf16* KL = state ? sKs : sKc;
    v8f acc = {};
#pragma unroll
    for (int k0 = 0; k0 < kC; k0 += 32) {
      v16bf a;   // A = K^T : A[m=d][k=t] = K_lds[t][d]
#pragma unroll
      for (int e = 0; e < 16; ++e) {
        int kk = k0 + 16 * half + e;
        a[e] = KL[kk * 72 + dt * 16 + l16];
      }
      v16bf bm = *(const v16bf*)(&sVt[(et * 16 + l16) * 80 + k0 + 16 * half]);
      acc = wmma_bf16(a, bm, acc);
    }
    float* dst = (state ? As : Ac) + (size_t)blk * (kDK * kDK);
#pragma unroll
    for (int ii = 0; ii < 8; ++ii) {
      int d = dt * 16 + ii + 8 * half;
      int e = et * 16 + l16;
      dst[d * kDK + e] = acc[ii];
    }
  }
  if (tid < 128) {                      // per-d column sums over the chunk
    int d = tid & 63;
    const __bf16* KL = (tid < 64) ? sKc : sKs;
    float s = 0.f;
#pragma unroll
    for (int t = 0; t < kC; ++t) s += bf2f(KL[t * 72 + d]);
    ((tid < 64) ? kcs : kss)[blk * kDK + d] = s;
  }
}

// ---------------- phase B: exclusive prefix over chunks ----------------
__global__ void k_scan(const float* __restrict__ Ac, const float* __restrict__ As,
                       const float* __restrict__ kcs, const float* __restrict__ kss,
                       __bf16* __restrict__ ScT, __bf16* __restrict__ SsT,
                       float* __restrict__ zc, float* __restrict__ zs) {
  const int bh = blockIdx.x;           // 16 blocks
  const int tid = threadIdx.x;         // 256 threads, 16 elems each
  float rc[16], rs[16];
#pragma unroll
  for (int j = 0; j < 16; ++j) { rc[j] = 0.f; rs[j] = 0.f; }
  for (int c = 0; c < kNC; ++c) {
    size_t base = ((size_t)bh * kNC + c) * (kDK * kDK);
#pragma unroll
    for (int j = 0; j < 16; ++j) {
      int li = tid * 16 + j;
      int d = li >> 6, e = li & 63;
      ScT[base + e * kDK + d] = f2bf(rc[j]);   // store S^T for WMMA B loads
      SsT[base + e * kDK + d] = f2bf(rs[j]);
      rc[j] += Ac[base + li];
      rs[j] += As[base + li];
    }
  }
  if (tid < 128) {
    int d = tid & 63;
    const float* src = (tid < 64) ? kcs : kss;
    float* dst = (tid < 64) ? zc : zs;
    float run = 0.f;
    for (int c = 0; c < kNC; ++c) {
      size_t i = ((size_t)bh * kNC + c) * kDK + d;
      dst[i] = run;
      run += src[i];
    }
  }
}

// ---------------- phase C: per-chunk output ----------------
__global__ void k_output(const __bf16* __restrict__ Qc, const __bf16* __restrict__ Qs,
                         const __bf16* __restrict__ Kc, const __bf16* __restrict__ Ks,
                         const __bf16* __restrict__ Vb,
                         const __bf16* __restrict__ ScT, const __bf16* __restrict__ SsT,
                         const float* __restrict__ zc, const float* __restrict__ zs,
                         const float* __restrict__ G, __bf16* __restrict__ Ob) {
  __shared__ __bf16 sP[kC * 72];       // masked P (bf16)
  __shared__ __bf16 sVt[kDK * 80];     // V transposed [e][t]
  __shared__ __bf16 sSc[kDK * 80];     // Sc^T chunk [e][d]
  __shared__ __bf16 sSs[kDK * 80];     // Ss^T chunk [e][d]
  __shared__ float  sZ[kC];
  const int blk = blockIdx.x;
  const int bh = blk >> 4, c = blk & 15;
  const int b = bh >> 3, h = bh & 7;
  const int row0 = b * kT + c * kC;
  const int col0 = h * kDK;
  const int tid = threadIdx.x;
  const size_t sbase = ((size_t)bh * kNC + c) * (kDK * kDK);

  // async-stage the S^T state chunks: each lane copies 16B global->LDS
  {
    const __bf16* scg = ScT + sbase;
    const __bf16* ssg = SsT + sbase;
#pragma unroll
    for (int i = 0; i < 2; ++i) {
      int ci = tid + i * 256;          // 0..511
      int e = ci >> 3, d8 = (ci & 7) * 8;
      unsigned lc = lds_off32(&sSc[e * 80 + d8]);
      unsigned ls = lds_off32(&sSs[e * 80 + d8]);
      unsigned long long gc = (unsigned long long)(scg + (size_t)e * kDK + d8);
      unsigned long long gs = (unsigned long long)(ssg + (size_t)e * kDK + d8);
      asm volatile("global_load_async_to_lds_b128 %0, %1, off"
                   :: "v"(lc), "v"(gc) : "memory");
      asm volatile("global_load_async_to_lds_b128 %0, %1, off"
                   :: "v"(ls), "v"(gs) : "memory");
    }
  }
  // V staged transposed (manual)
#pragma unroll
  for (int i = 0; i < 16; ++i) {
    int li = tid + i * 256;
    int t = li >> 6, e = li & 63;
    sVt[e * 80 + t] = Vb[(size_t)(row0 + t) * kD + col0 + e];
  }
  asm volatile("s_wait_asynccnt 0x0" ::: "memory");
  __syncthreads();

  const int wave = tid >> 5, lane = tid & 31;
  const int half = lane >> 4, l16 = lane & 15;

  // intra-chunk scores P with causal mask
  for (int j = wave; j < 16; j += 8) {
    int tm = j >> 2, tn = j & 3;
    const __bf16* qc  = Qc + (size_t)(row0 + tm * 16 + l16) * kD + col0;
    const __bf16* qs  = Qs + (size_t)(row0 + tm * 16 + l16) * kD + col0;
    const __bf16* kcb = Kc + (size_t)(row0 + tn * 16 + l16) * kD + col0;
    const __bf16* ksb = Ks + (size_t)(row0 + tn * 16 + l16) * kD + col0;
    v8f p = {};
#pragma unroll
    for (int k0 = 0; k0 < kDK; k0 += 32) {
      p = wmma_bf16(load_a16(qc + k0, half), *(const v16bf*)(kcb + k0 + 16 * half), p);
      p = wmma_bf16(load_a16(qs + k0, half), *(const v16bf*)(ksb + k0 + 16 * half), p);
    }
#pragma unroll
    for (int ii = 0; ii < 8; ++ii) {
      int t = tm * 16 + ii + 8 * half;
      int s = tn * 16 + l16;
      sP[t * 72 + s] = f2bf(s <= t ? p[ii] : 0.f);
    }
  }
  __syncthreads();

  // normalizer: z[t] = Qc.zc_prev + Qs.zs_prev + rowsum(masked P)
  if (tid < kC) {
    int t = tid;
    const __bf16* qc = Qc + (size_t)(row0 + t) * kD + col0;
    const __bf16* qs = Qs + (size_t)(row0 + t) * kD + col0;
    const float* zcp = zc + ((size_t)bh * kNC + c) * kDK;
    const float* zsp = zs + ((size_t)bh * kNC + c) * kDK;
    float acc = 0.f;
#pragma unroll 8
    for (int d = 0; d < kDK; ++d)
      acc += bf2f(qc[d]) * zcp[d] + bf2f(qs[d]) * zsp[d];
#pragma unroll 8
    for (int s = 0; s < kC; ++s) acc += bf2f(sP[t * 72 + s]);
    sZ[t] = fmaxf(acc, 1e-6f);
  }
  __syncthreads();

  // o = Qc@Sc_prev + Qs@Ss_prev + P@V ; normalize, gate, store bf16
  for (int j = wave; j < 16; j += 8) {
    int tm = j >> 2, tn = j & 3;
    const __bf16* qc = Qc + (size_t)(row0 + tm * 16 + l16) * kD + col0;
    const __bf16* qs = Qs + (size_t)(row0 + tm * 16 + l16) * kD + col0;
    const int ecol = tn * 16 + l16;
    v8f o = {};
#pragma unroll
    for (int k0 = 0; k0 < kDK; k0 += 32) {
      o = wmma_bf16(load_a16(qc + k0, half),
                    *(const v16bf*)(&sSc[ecol * 80 + k0 + 16 * half]), o);
      o = wmma_bf16(load_a16(qs + k0, half),
                    *(const v16bf*)(&sSs[ecol * 80 + k0 + 16 * half]), o);
      o = wmma_bf16(load_a16(&sP[(tm * 16 + l16) * 72 + k0], half),
                    *(const v16bf*)(&sVt[ecol * 80 + k0 + 16 * half]), o);
    }
#pragma unroll
    for (int ii = 0; ii < 8; ++ii) {
      int t = tm * 16 + ii + 8 * half;
      int e = tn * 16 + l16;
      size_t g = (size_t)(row0 + t) * kD + col0 + e;
      Ob[g] = f2bf(o[ii] / sZ[t] * G[g]);
    }
  }
}

// ---------------- final GEMM: out = Ob @ Wo, 16x64 tile per wave, double-buffered ----------------
__global__ void k_final(const __bf16* __restrict__ Ob, const __bf16* __restrict__ WoT,
                        float* __restrict__ out) {
  const int wave = threadIdx.x >> 5, lane = threadIdx.x & 31;
  const int job = blockIdx.x * 8 + wave;   // 0..1023: 128 x 8
  const int tm = job >> 3, tg = job & 7;
  const int half = lane >> 4, l16 = lane & 15;
  const __bf16* aptr = Ob  + (size_t)(tm * 16 + l16) * kD;
  const __bf16* bptr = WoT + (size_t)(tg * 64 + l16) * kD + 16 * half;
  v8f acc[4] = {};
  v16bf a0 = load_a16(aptr, half);
  v16bf b0[4];
#pragma unroll
  for (int t4 = 0; t4 < 4; ++t4)
    b0[t4] = *(const v16bf*)(bptr + (size_t)t4 * 16 * kD);
#pragma unroll
  for (int k0 = 32; k0 < kD; k0 += 32) {
    v16bf a1 = load_a16(aptr + k0, half);
    v16bf b1[4];
#pragma unroll
    for (int t4 = 0; t4 < 4; ++t4)
      b1[t4] = *(const v16bf*)(bptr + (size_t)t4 * 16 * kD + k0);
#pragma unroll
    for (int t4 = 0; t4 < 4; ++t4)
      acc[t4] = wmma_bf16(a0, b0[t4], acc[t4]);
    a0 = a1;
#pragma unroll
    for (int t4 = 0; t4 < 4; ++t4) b0[t4] = b1[t4];
  }
#pragma unroll
  for (int t4 = 0; t4 < 4; ++t4)
    acc[t4] = wmma_bf16(a0, b0[t4], acc[t4]);
#pragma unroll
  for (int ii = 0; ii < 8; ++ii) {
    int r = tm * 16 + ii + 8 * half;
#pragma unroll
    for (int t4 = 0; t4 < 4; ++t4)
      out[(size_t)r * kD + tg * 64 + t4 * 16 + l16] = acc[t4][ii];
  }
}

extern "C" void kernel_launch(void* const* d_in, const int* in_sizes, int n_in,
                              void* d_out, int out_size, void* d_ws, size_t ws_size,
                              hipStream_t stream) {
  const float* x  = (const float*)d_in[0];
  const float* Wq = (const float*)d_in[1];
  const float* Wk = (const float*)d_in[2];
  const float* Wv = (const float*)d_in[3];
  const float* Wo = (const float*)d_in[4];
  const float* Wg = (const float*)d_in[5];
  const float* bg = (const float*)d_in[6];

  char* ws = (char*)d_ws;
  size_t off = 0;
  auto alloc = [&](size_t bytes) -> void* {
    void* p = ws + off;
    off = (off + bytes + 255) & ~(size_t)255;
    return p;
  };
  __bf16* xb  = (__bf16*)alloc((size_t)kN * kD * 2);
  __bf16* WqT = (__bf16*)alloc((size_t)kD * kD * 2);
  __bf16* WkT = (__bf16*)alloc((size_t)kD * kD * 2);
  __bf16* WvT = (__bf16*)alloc((size_t)kD * kD * 2);
  __bf16* WgT = (__bf16*)alloc((size_t)kD * kD * 2);
  __bf16* WoT = (__bf16*)alloc((size_t)kD * kD * 2);
  __bf16* Qc  = (__bf16*)alloc((size_t)kN * kD * 2);
  __bf16* Qs  = (__bf16*)alloc((size_t)kN * kD * 2);
  __bf16* Kc  = (__bf16*)alloc((size_t)kN * kD * 2);
  __bf16* Ks  = (__bf16*)alloc((size_t)kN * kD * 2);
  __bf16* Vb  = (__bf16*)alloc((size_t)kN * kD * 2);
  float*  G   = (float*) alloc((size_t)kN * kD * 4);
  float*  Ac  = (float*) alloc((size_t)kBH * kNC * kDK * kDK * 4);
  float*  As  = (float*) alloc((size_t)kBH * kNC * kDK * kDK * 4);
  float*  kcs = (float*) alloc((size_t)kBH * kNC * kDK * 4);
  float*  kss = (float*) alloc((size_t)kBH * kNC * kDK * 4);
  __bf16* ScT = (__bf16*)alloc((size_t)kBH * kNC * kDK * kDK * 2);
  __bf16* SsT = (__bf16*)alloc((size_t)kBH * kNC * kDK * kDK * 2);
  float*  zcp = (float*) alloc((size_t)kBH * kNC * kDK * 4);
  float*  zsp = (float*) alloc((size_t)kBH * kNC * kDK * 4);
  __bf16* Ob  = (__bf16*)alloc((size_t)kN * kD * 2);

  k_conv_x<<<(kN * kD) / 256, 256, 0, stream>>>(x, xb, kN * kD);
  dim3 gw((kD * kD) / 256, 5);
  k_conv_wT<<<gw, 256, 0, stream>>>(Wq, Wk, Wv, Wg, Wo, WqT, WkT, WvT, WgT, WoT);
  k_proj<<<512, 256, 0, stream>>>(xb, WqT, WkT, WvT, WgT, bg, Qc, Qs, Kc, Ks, Vb, G);
  k_chunk_kv<<<kBH * kNC, 256, 0, stream>>>(Kc, Ks, Vb, Ac, As, kcs, kss);
  k_scan<<<kBH, 256, 0, stream>>>(Ac, As, kcs, kss, ScT, SsT, zcp, zsp);
  k_output<<<kBH * kNC, 256, 0, stream>>>(Qc, Qs, Kc, Ks, Vb, ScT, SsT, zcp, zsp, G, Ob);
  k_final<<<128, 256, 0, stream>>>(Ob, WoT, (float*)d_out);
}
